// SurvivalGNN_5317169513097
// MI455X (gfx1250) — compile-verified, matrix operand order checked by reading, not run on hardware
//
#include <hip/hip_runtime.h>
#include <hip/hip_bf16.h>

typedef __attribute__((ext_vector_type(16))) _Float16 v16h;
typedef __attribute__((ext_vector_type(8)))  float    v8f;

// ---------------------------------------------------------------------------
// Zero-fill
// ---------------------------------------------------------------------------
__global__ __launch_bounds__(256) void zero_f32(float* __restrict__ p, int n) {
    int i = blockIdx.x * blockDim.x + threadIdx.x;
    if (i < n) p[i] = 0.0f;
}

// ---------------------------------------------------------------------------
// Degree count: cnt[dst[e]] += 1
// ---------------------------------------------------------------------------
__global__ __launch_bounds__(256) void count_edges(const long long* __restrict__ dst,
                                                   float* __restrict__ cnt, int E) {
    int e = blockIdx.x * blockDim.x + threadIdx.x;
    if (e < E) atomicAdd(cnt + (int)dst[e], 1.0f);
}

// ---------------------------------------------------------------------------
// Fused dual GEMM:  Y = X @ Wl.T,  Z = X @ Wr.T   (Wl, Wr are [64,K] f32).
// K is a compile-time constant (128 or 64) so the K-loop fully unrolls:
// K/32 iterations x 2 WMMAs, with the A fragment converted once per step.
// One wave owns one 16x16 tile of both outputs; block = 8 waves = 32 rows.
// A layout (16-bit 16x32, ISA 7.12.2): lane<16 holds K=0..7,16..23 of row
// lane; lane>=16 holds K=8..15,24..31 of row lane-16. B mirrors with N.
// ---------------------------------------------------------------------------
template <int K>
__global__ __launch_bounds__(256) void gemm_dual_wmma(const float* __restrict__ X,
                                                      const float* __restrict__ Wl,
                                                      const float* __restrict__ Wr,
                                                      float* __restrict__ Y,
                                                      float* __restrict__ Z,
                                                      int N) {
    const int lane = threadIdx.x & 31;
    const int wave = threadIdx.x >> 5;      // 0..7
    const int rowBase = blockIdx.x * 32 + (wave >> 2) * 16;
    const int colBase = (wave & 3) * 16;

    const int half = lane >> 4;             // 0 or 1
    const int l16  = lane & 15;

    int row = rowBase + l16;
    if (row >= N) row = N - 1;              // clamp: EXEC stays all-ones at WMMA
    const int wrow = colBase + l16;         // weight row == output feature

    v8f accL = {};
    v8f accR = {};
#pragma unroll
    for (int k0 = 0; k0 < K; k0 += 32) {
        const float* xp  = X  + (long long)row  * K + k0 + half * 8;
        const float* wlp = Wl + (long long)wrow * K + k0 + half * 8;
        const float* wrp = Wr + (long long)wrow * K + k0 + half * 8;
        v16h a, bl, br;
#pragma unroll
        for (int e = 0; e < 8; ++e) {
            a[e]      = (_Float16)xp[e];
            a[e + 8]  = (_Float16)xp[e + 16];
            bl[e]     = (_Float16)wlp[e];
            bl[e + 8] = (_Float16)wlp[e + 16];
            br[e]     = (_Float16)wrp[e];
            br[e + 8] = (_Float16)wrp[e + 16];
        }
        accL = __builtin_amdgcn_wmma_f32_16x16x32_f16(
            false, a, false, bl, (short)0, accL, false, false);
        accR = __builtin_amdgcn_wmma_f32_16x16x32_f16(
            false, a, false, br, (short)0, accR, false, false);
    }

    // D layout: lane half 0 -> rows rowBase+v, half 1 -> rows rowBase+8+v;
    // col = colBase + l16.
    const int r0  = rowBase + half * 8;
    const int col = colBase + l16;
    if (r0 + 8 <= N) {                      // full tile (always true: N % 32 == 0)
#pragma unroll
        for (int v = 0; v < 8; ++v) {
            long long o = (long long)(r0 + v) * 64 + col;
            Y[o] = accL[v];
            Z[o] = accR[v];
        }
    } else {
#pragma unroll
        for (int v = 0; v < 8; ++v) {
            if (r0 + v < N) {
                long long o = (long long)(r0 + v) * 64 + col;
                Y[o] = accL[v];
                Z[o] = accR[v];
            }
        }
    }
}

// ---------------------------------------------------------------------------
// Scatter-add 64-dim rows: S[dst[e]] += Y[src[e]]. One thread = one float4
// chunk of one edge (E*16 threads). Coalesced gather + contiguous atomics.
// ---------------------------------------------------------------------------
__global__ __launch_bounds__(256) void scatter_add64(const float* __restrict__ Y,
                                                     const long long* __restrict__ src,
                                                     const long long* __restrict__ dst,
                                                     float* __restrict__ S, int E) {
    long long tid = (long long)blockIdx.x * blockDim.x + threadIdx.x;
    int e = (int)(tid >> 4);
    if (e >= E) return;
    int c = (int)(tid & 15) * 4;
    int s = (int)src[e];
    int d = (int)dst[e];
    const float4 v = *(const float4*)(Y + (long long)s * 64 + c);
    float* p = S + (long long)d * 64 + c;
    atomicAdd(p + 0, v.x);
    atomicAdd(p + 1, v.y);
    atomicAdd(p + 2, v.z);
    atomicAdd(p + 3, v.w);
}

// ---------------------------------------------------------------------------
// H[i] = relu(S[i]/max(cnt,1) + bias[f] + Z[i])   (safe when H aliases S or Z)
// ---------------------------------------------------------------------------
__global__ __launch_bounds__(256) void combine_relu(const float* __restrict__ S,
                                                    const float* __restrict__ cnt,
                                                    const float* __restrict__ bias,
                                                    const float* __restrict__ Z,
                                                    float* __restrict__ H, int total) {
    int i = blockIdx.x * blockDim.x + threadIdx.x;
    if (i >= total) return;
    int n = i >> 6, f = i & 63;
    float c = cnt[n];
    c = c > 1.0f ? c : 1.0f;
    float h = S[i] / c + bias[f] + Z[i];
    H[i] = h > 0.0f ? h : 0.0f;
}

// ---------------------------------------------------------------------------
// logits[n, 0..4] = bh + Wh @ h2[n]   (Wh is [5,64]); weights staged in LDS,
// one thread per node, float4 row reads.
// ---------------------------------------------------------------------------
__global__ __launch_bounds__(256) void head_kernel(const float* __restrict__ H,
                                                   const float* __restrict__ Wh,
                                                   const float* __restrict__ bh,
                                                   float* __restrict__ L, int N) {
    __shared__ float w[5 * 64];
    __shared__ float b[5];
    for (int i = threadIdx.x; i < 5 * 64; i += blockDim.x) w[i] = Wh[i];
    if (threadIdx.x < 5) b[threadIdx.x] = bh[threadIdx.x];
    __syncthreads();

    int n = blockIdx.x * blockDim.x + threadIdx.x;
    if (n >= N) return;
    float acc[5];
#pragma unroll
    for (int o = 0; o < 5; ++o) acc[o] = b[o];
    const float* h = H + (long long)n * 64;
#pragma unroll
    for (int f = 0; f < 64; f += 4) {
        float4 hv = *(const float4*)(h + f);
#pragma unroll
        for (int o = 0; o < 5; ++o) {
            acc[o] += hv.x * w[o * 64 + f]     + hv.y * w[o * 64 + f + 1]
                    + hv.z * w[o * 64 + f + 2] + hv.w * w[o * 64 + f + 3];
        }
    }
#pragma unroll
    for (int o = 0; o < 5; ++o) L[(long long)n * 5 + o] = acc[o];
}

// ---------------------------------------------------------------------------
extern "C" void kernel_launch(void* const* d_in, const int* in_sizes, int n_in,
                              void* d_out, int out_size, void* d_ws, size_t ws_size,
                              hipStream_t stream) {
    const float*     x   = (const float*)d_in[0];
    const long long* ei  = (const long long*)d_in[1];
    const float*     W1l = (const float*)d_in[2];
    const float*     b1l = (const float*)d_in[3];
    const float*     W1r = (const float*)d_in[4];
    const float*     W2l = (const float*)d_in[5];
    const float*     b2l = (const float*)d_in[6];
    const float*     W2r = (const float*)d_in[7];
    const float*     Wh  = (const float*)d_in[8];
    const float*     bh  = (const float*)d_in[9];

    const int N = in_sizes[0] / 128;        // 100000
    const int E = in_sizes[1] / 2;          // 3200000
    const long long* src = ei;
    const long long* dst = ei + E;

    float* out_logits = (float*)d_out;                      // N*5
    float* out_h2     = (float*)d_out + (long long)N * 5;   // N*64

    // workspace: cnt | bufA | bufB | bufC   (N + 3*N*64 floats ~ 77 MB)
    float* cnt  = (float*)d_ws;
    float* bufA = cnt + N;
    float* bufB = bufA + (long long)N * 64;
    float* bufC = bufB + (long long)N * 64;

    const int NH = N * 64;
    const dim3 blk(256);
    const int gZeroN  = (N + 255) / 256;
    const int gZeroNH = (NH + 255) / 256;
    const int gEdge   = (E + 255) / 256;
    const long long scatterThreads = (long long)E * 16;
    const int gScat   = (int)((scatterThreads + 255) / 256);
    const int gGemm   = (N + 31) / 32;      // 32 rows per block

    // degree counts (shared by both layers)
    zero_f32<<<gZeroN, blk, 0, stream>>>(cnt, N);
    count_edges<<<gEdge, blk, 0, stream>>>(dst, cnt, E);

    // ---- layer 1: project-then-aggregate (halves edge traffic vs reference);
    //      fused dual GEMM reads x once for both Wl and Wr projections.
    gemm_dual_wmma<128><<<gGemm, blk, 0, stream>>>(x, W1l, W1r, bufA, bufC, N); // y1, z1
    zero_f32<<<gZeroNH, blk, 0, stream>>>(bufB, NH);
    scatter_add64<<<gScat, blk, 0, stream>>>(bufA, src, dst, bufB, E);          // s1
    combine_relu<<<gZeroNH, blk, 0, stream>>>(bufB, cnt, b1l, bufC, bufB, NH);  // h1 in bufB

    // ---- layer 2 (z2 parked in the output h2 region; combine is in-place)
    gemm_dual_wmma<64><<<gGemm, blk, 0, stream>>>(bufB, W2l, W2r, bufA, out_h2, N); // y2, z2
    zero_f32<<<gZeroNH, blk, 0, stream>>>(bufC, NH);
    scatter_add64<<<gScat, blk, 0, stream>>>(bufA, src, dst, bufC, E);              // s2
    combine_relu<<<gZeroNH, blk, 0, stream>>>(bufC, cnt, b2l, out_h2, out_h2, NH);  // h2

    // ---- head
    head_kernel<<<gZeroN, blk, 0, stream>>>(out_h2, Wh, bh, out_logits, N);

    (void)n_in; (void)out_size; (void)ws_size;
}
